// GPTLanguageModel_41120016892815
// MI455X (gfx1250) — matrix operand order repaired and checked by reading, not compiled, not run on hardware
//
#include <hip/hip_runtime.h>
#include <hip/hip_bf16.h>

// ---- model constants -------------------------------------------------------
static constexpr int Vv  = 50257;
static constexpr int Dd  = 768;
static constexpr int Hh  = 12;
static constexpr int HS  = 64;
static constexpr int Lc  = 6;
static constexpr int Bc  = 2;
static constexpr int Tc  = 2048;
static constexpr int BT  = Bc * Tc;   // 4096
static constexpr int D4  = 4 * Dd;    // 3072

typedef __attribute__((ext_vector_type(16))) _Float16 v16h;
typedef __attribute__((ext_vector_type(8)))  float    v8f;

// ---------------------------------------------------------------------------
// Embedding: x[bt,d] = tok_emb[idx[bt], d] + pos_emb[bt % T, d]
// ---------------------------------------------------------------------------
__global__ __launch_bounds__(256)
void embed_kernel(const int* __restrict__ idx, const float* __restrict__ tok,
                  const float* __restrict__ pos, float* __restrict__ x)
{
    int bt = blockIdx.x;
    int token = idx[bt];
    int t = bt % Tc;
    for (int d = threadIdx.x; d < Dd; d += 256)
        x[(size_t)bt * Dd + d] = tok[(size_t)token * Dd + d] + pos[(size_t)t * Dd + d];
}

// ---------------------------------------------------------------------------
// LayerNorm: one 256-thread block per row of D=768 (3 elems/thread)
// ---------------------------------------------------------------------------
__global__ __launch_bounds__(256)
void ln_kernel(const float* __restrict__ x, const float* __restrict__ gamma,
               const float* __restrict__ beta, float* __restrict__ out)
{
    int row = blockIdx.x, tid = threadIdx.x;
    const float* xr = x + (size_t)row * Dd;
    float v0 = xr[tid], v1 = xr[tid + 256], v2 = xr[tid + 512];

    __shared__ float red[256];
    red[tid] = v0 + v1 + v2;
    __syncthreads();
    for (int off = 128; off; off >>= 1) {
        if (tid < off) red[tid] += red[tid + off];
        __syncthreads();
    }
    float mean = red[0] * (1.0f / Dd);
    __syncthreads();

    float d0 = v0 - mean, d1 = v1 - mean, d2 = v2 - mean;
    red[tid] = d0 * d0 + d1 * d1 + d2 * d2;
    __syncthreads();
    for (int off = 128; off; off >>= 1) {
        if (tid < off) red[tid] += red[tid + off];
        __syncthreads();
    }
    float inv = rsqrtf(red[0] * (1.0f / Dd) + 1e-5f);

    float* orow = out + (size_t)row * Dd;
    orow[tid]       = d0 * inv * gamma[tid]       + beta[tid];
    orow[tid + 256] = d1 * inv * gamma[tid + 256] + beta[tid + 256];
    orow[tid + 512] = d2 * inv * gamma[tid + 512] + beta[tid + 512];
}

// ---------------------------------------------------------------------------
// WMMA fragment loaders (contiguous k order: k = 16*half + s), f32 -> f16 fragments.
// ---------------------------------------------------------------------------
__device__ __forceinline__ v16h load_a_frag(const float* __restrict__ ap)
{
    // lane holds A[row][aoff + 0..15]
    float4 x0 = *(const float4*)(ap);
    float4 x1 = *(const float4*)(ap + 4);
    float4 x2 = *(const float4*)(ap + 8);
    float4 x3 = *(const float4*)(ap + 12);
    v16h f;
    f[0]=(_Float16)x0.x;  f[1]=(_Float16)x0.y;  f[2]=(_Float16)x0.z;  f[3]=(_Float16)x0.w;
    f[4]=(_Float16)x1.x;  f[5]=(_Float16)x1.y;  f[6]=(_Float16)x1.z;  f[7]=(_Float16)x1.w;
    f[8]=(_Float16)x2.x;  f[9]=(_Float16)x2.y;  f[10]=(_Float16)x2.z; f[11]=(_Float16)x2.w;
    f[12]=(_Float16)x3.x; f[13]=(_Float16)x3.y; f[14]=(_Float16)x3.z; f[15]=(_Float16)x3.w;
    return f;
}

// 16 consecutive floats -> fragment in element order (used for K^T B-frags:
// lane = key column, 16 contiguous head-dim values per lane).
__device__ __forceinline__ v16h load_contig_frag(const float* __restrict__ p)
{
    float4 x0 = *(const float4*)(p);
    float4 x1 = *(const float4*)(p + 4);
    float4 x2 = *(const float4*)(p + 8);
    float4 x3 = *(const float4*)(p + 12);
    v16h f;
    f[0]=(_Float16)x0.x;  f[1]=(_Float16)x0.y;  f[2]=(_Float16)x0.z;  f[3]=(_Float16)x0.w;
    f[4]=(_Float16)x1.x;  f[5]=(_Float16)x1.y;  f[6]=(_Float16)x1.z;  f[7]=(_Float16)x1.w;
    f[8]=(_Float16)x2.x;  f[9]=(_Float16)x2.y;  f[10]=(_Float16)x2.z; f[11]=(_Float16)x2.w;
    f[12]=(_Float16)x3.x; f[13]=(_Float16)x3.y; f[14]=(_Float16)x3.z; f[15]=(_Float16)x3.w;
    return f;
}

// B fragment out of the LDS tile (32 K-rows x 64 N-cols, f32 row-major).
__device__ __forceinline__ v16h load_b_frag_lds(const float* bp)
{
    v16h f;
    #pragma unroll
    for (int v = 0; v < 8; ++v) {
        f[2 * v]     = (_Float16)bp[(2 * v) * 64];
        f[2 * v + 1] = (_Float16)bp[(2 * v + 1) * 64];
    }
    return f;
}

// B fragment from V (32 key-rows x 64 e-cols, row stride 64):
// vp = &V[s0 + boff][e0 + l16]; f[j] = V[s0 + boff + j][e0 + l16].
__device__ __forceinline__ v16h load_bV_frag(const float* __restrict__ vp)
{
    v16h f;
    #pragma unroll
    for (int j = 0; j < 16; ++j) f[j] = (_Float16)vp[j * 64];
    return f;
}

__device__ __forceinline__ float rowmax16(float x)
{
    #pragma unroll
    for (int m = 8; m; m >>= 1) x = fmaxf(x, __shfl_xor(x, m, 32));
    return x;
}
__device__ __forceinline__ float rowsum16(float x)
{
    #pragma unroll
    for (int m = 8; m; m >>= 1) x += __shfl_xor(x, m, 32);
    return x;
}

// ---------------------------------------------------------------------------
// WMMA GEMM:  C = A @ W (+bias) (+residual) (ReLU?)      f16 MACs, f32 acc
// Block = 8 waves as 4(M) x 2(N); wave owns a 32x32 C tile (4 v_wmma/step).
// B tile (32K x 64N f32, 8KB) staged to LDS, double-buffered via gfx1250
// async global->LDS copies (ASYNCcnt), shared by all 8 waves.
// ---------------------------------------------------------------------------
template<int N, int K, bool RELU>
__global__ __launch_bounds__(256)
void gemm_wmma(const float* __restrict__ A, const float* __restrict__ W,
               float* C, const float* __restrict__ bias,
               const float* residual,
               long long wBatchStride, long long cBatchStride)
{
    constexpr bool RAGGED = (N % 64) != 0;
    constexpr int  PEND   = RAGGED ? 8 : 2;   // async loads issued per stage

    __shared__ __align__(16) float btile[2][32 * 64];

    const int tid  = threadIdx.x;
    const int wave = tid >> 5;
    const int lane = tid & 31;
    const int wm = wave & 3;          // 0..3  -> M sub-tile
    const int wn = wave >> 2;         // 0..1  -> N sub-tile
    const int m0 = blockIdx.y * 128 + wm * 32;
    const int n0base = blockIdx.x * 64;
    const int n0 = n0base + wn * 32;

    const float* Wb = W + (long long)blockIdx.z * wBatchStride;
    float*       Cb = C + (long long)blockIdx.z * cBatchStride;

    const int half = lane >> 4;       // half-wave select
    const int l16  = lane & 15;
    const int aoff = half * 16;       // A-frag K offset for lanes 16-31
    const int boff = half * 16;       // B-frag K offset for lanes 16-31

    const int r0 = m0 + l16;          // A rows for the two M tiles
    const int col0 = n0 + l16;        // C columns for the two N tiles
    const int col1 = col0 + 16;

    const float* apA = A + (size_t)r0 * K + aoff;
    const float* apB = apA + (size_t)16 * K;

    auto stage = [&](int buf, int k0) {
        if constexpr (!RAGGED) {
            #pragma unroll
            for (int j = 0; j < 2; ++j) {          // 512 float4 / 256 threads
                int idx4 = tid + j * 256;
                int row  = idx4 >> 4;
                int col  = (idx4 & 15) * 4;
                const float* gp = Wb + (size_t)(k0 + row) * N + n0base + col;
                unsigned la = (unsigned)(size_t)&btile[buf][row * 64 + col];
                asm volatile("global_load_async_to_lds_b128 %0, %1, off"
                             :: "v"(la), "v"(gp) : "memory");
            }
        } else {
            #pragma unroll
            for (int j = 0; j < 8; ++j) {          // 2048 floats / 256 threads
                int idx = tid + j * 256;
                int row = idx >> 6;
                int col = idx & 63;
                int gc  = n0base + col;
                gc = (gc < N) ? gc : (N - 1);      // clamp: dup columns only
                const float* gp = Wb + (size_t)(k0 + row) * N + gc;
                unsigned la = (unsigned)(size_t)&btile[buf][row * 64 + col];
                asm volatile("global_load_async_to_lds_b32 %0, %1, off"
                             :: "v"(la), "v"(gp) : "memory");
            }
        }
    };

    v8f acc00 = {0.f,0.f,0.f,0.f,0.f,0.f,0.f,0.f};
    v8f acc01 = acc00, acc10 = acc00, acc11 = acc00;

    stage(0, 0);

    for (int k0 = 0; k0 < K; k0 += 32) {
        const int cur = (k0 >> 5) & 1;
        if (k0 + 32 < K) {
            stage(cur ^ 1, k0 + 32);               // prefetch next tile
            asm volatile("s_wait_asynccnt %0" :: "n"(PEND));
        } else {
            asm volatile("s_wait_asynccnt 0x0");
        }
        __syncthreads();

        v16h a0 = load_a_frag(apA);
        v16h a1 = load_a_frag(apB);
        const float* bb = &btile[cur][boff * 64 + wn * 32 + l16];
        v16h b0 = load_b_frag_lds(bb);
        v16h b1 = load_b_frag_lds(bb + 16);

        acc00 = __builtin_amdgcn_wmma_f32_16x16x32_f16(false, a0, false, b0, (short)0, acc00, false, false);
        acc01 = __builtin_amdgcn_wmma_f32_16x16x32_f16(false, a0, false, b1, (short)0, acc01, false, false);
        acc10 = __builtin_amdgcn_wmma_f32_16x16x32_f16(false, a1, false, b0, (short)0, acc10, false, false);
        acc11 = __builtin_amdgcn_wmma_f32_16x16x32_f16(false, a1, false, b1, (short)0, acc11, false, false);

        apA += 32; apB += 32;
        __syncthreads();
    }

    const int mb0 = m0 + half * 8;
    const int mb1 = m0 + 16 + half * 8;
    auto store_tile = [&](const v8f& acc, int mbase, int col) {
        if (RAGGED && col >= N) return;
        float bv = bias ? bias[col] : 0.f;
        #pragma unroll
        for (int v = 0; v < 8; ++v) {
            size_t ci = (size_t)(mbase + v) * N + col;
            float val = acc[v] + bv;
            if (residual) val += residual[ci];
            if (RELU) val = fmaxf(val, 0.f);
            Cb[ci] = val;
        }
    };
    store_tile(acc00, mb0, col0);
    store_tile(acc01, mb0, col1);
    store_tile(acc10, mb1, col0);
    store_tile(acc11, mb1, col1);
}

// ---------------------------------------------------------------------------
// WMMA flash attention (causal, online softmax).
// One wave per 16-query tile; 32-key chunks:
//   S (16x32)  = Q(16x64) . K^T        -> 4 WMMAs (2 key subtiles x 2 halves)
//   softmax in C layout: row = (half, vgpr); 16-lane shuffle reductions
//   P transposed C->A layout through a 2KB per-wave LDS tile (DS in-order)
//   O (16x64) += P(16x32) . V(32x64)   -> 4 WMMAs (4 e-subtiles)
// q/k/v layout: [h][b*T + t][HS]; output: ocat[bt][h*64 + e].
// ---------------------------------------------------------------------------
__global__ __launch_bounds__(128)
void attn_wmma_kernel(const float* __restrict__ q, const float* __restrict__ k,
                      const float* __restrict__ v, float* __restrict__ ocat)
{
    __shared__ __align__(16) float ptile[4][16 * 32];

    const int wv_  = threadIdx.y;                 // wave in block (0..3)
    const int lane = threadIdx.x;                 // 0..31
    const int qt   = blockIdx.x * 4 + wv_;        // 16-query tile index
    const int h = blockIdx.y, b = blockIdx.z;

    const int half = lane >> 4, l16 = lane & 15;
    const int aoff = half * 16;
    const int boff = half * 16;

    const size_t head_base = ((size_t)h * BT + (size_t)b * Tc) * HS;
    const float* qbse = q + head_base;
    const float* kbse = k + head_base;
    const float* vbse = v + head_base;

    const int qrow = qt * 16 + l16;               // A-frag row (query)
    v16h aq0 = load_a_frag(qbse + (size_t)qrow * HS + aoff);        // k 0..31
    v16h aq1 = load_a_frag(qbse + (size_t)qrow * HS + 32 + aoff);   // k 32..63

    v8f zero = {0.f,0.f,0.f,0.f,0.f,0.f,0.f,0.f};
    v8f o0 = zero, o1 = zero, o2 = zero, o3 = zero;   // O 16x64 (4 e-subtiles)
    float mrow[8], lrow[8];
    #pragma unroll
    for (int e = 0; e < 8; ++e) { mrow[e] = -1e30f; lrow[e] = 0.f; }

    const int crow = qt * 16 + half * 8;          // C-layout row base
    const int qmax = qt * 16 + 15;
    const float scale = 0.125f;                   // 1/sqrt(64)
    float* pt = &ptile[wv_][0];

    for (int s0 = 0; s0 <= qmax; s0 += 32) {
        // ---- S = Q . K^T ----------------------------------------------------
        const float* kp = kbse + (size_t)(s0 + l16) * HS + boff;
        v8f sA = zero, sB = zero;
        {
            v16h b0 = load_contig_frag(kp);
            v16h b1 = load_contig_frag(kp + 32);
            sA = __builtin_amdgcn_wmma_f32_16x16x32_f16(false, aq0, false, b0, (short)0, sA, false, false);
            sA = __builtin_amdgcn_wmma_f32_16x16x32_f16(false, aq1, false, b1, (short)0, sA, false, false);
            const float* kp2 = kp + 16 * HS;
            v16h b2 = load_contig_frag(kp2);
            v16h b3 = load_contig_frag(kp2 + 32);
            sB = __builtin_amdgcn_wmma_f32_16x16x32_f16(false, aq0, false, b2, (short)0, sB, false, false);
            sB = __builtin_amdgcn_wmma_f32_16x16x32_f16(false, aq1, false, b3, (short)0, sB, false, false);
        }
        // ---- online softmax (C layout) -------------------------------------
        #pragma unroll
        for (int e = 0; e < 8; ++e) {
            int row = crow + e;
            float sa = (s0 + l16      <= row) ? sA[e] * scale : -1e30f;
            float sb = (s0 + 16 + l16 <= row) ? sB[e] * scale : -1e30f;
            float cm = rowmax16(fmaxf(sa, sb));
            float mn = fmaxf(mrow[e], cm);
            float corr = __expf(mrow[e] - mn);
            float pa = __expf(sa - mn);
            float pb = __expf(sb - mn);
            lrow[e] = lrow[e] * corr + rowsum16(pa + pb);
            mrow[e] = mn;
            o0[e] *= corr; o1[e] *= corr; o2[e] *= corr; o3[e] *= corr;
            pt[(half * 8 + e) * 32 + l16]      = pa;   // transpose via LDS
            pt[(half * 8 + e) * 32 + 16 + l16] = pb;
        }
        // ---- P A-frag from LDS (same-wave DS ops are in-order) -------------
        v16h ap;
        {
            const float* pr = pt + l16 * 32 + aoff;
            #pragma unroll
            for (int j = 0; j < 8; ++j) ap[j]     = (_Float16)pr[j];
            #pragma unroll
            for (int j = 0; j < 8; ++j) ap[8 + j] = (_Float16)pr[8 + j];
        }
        // ---- O += P . V -----------------------------------------------------
        const float* vp = vbse + (size_t)(s0 + boff) * HS + l16;
        o0 = __builtin_amdgcn_wmma_f32_16x16x32_f16(false, ap, false, load_bV_frag(vp),      (short)0, o0, false, false);
        o1 = __builtin_amdgcn_wmma_f32_16x16x32_f16(false, ap, false, load_bV_frag(vp + 16), (short)0, o1, false, false);
        o2 = __builtin_amdgcn_wmma_f32_16x16x32_f16(false, ap, false, load_bV_frag(vp + 32), (short)0, o2, false, false);
        o3 = __builtin_amdgcn_wmma_f32_16x16x32_f16(false, ap, false, load_bV_frag(vp + 48), (short)0, o3, false, false);
    }

    // ---- normalize and store O ---------------------------------------------
    float invl[8];
    #pragma unroll
    for (int e = 0; e < 8; ++e) invl[e] = 1.0f / lrow[e];
    float* obase = ocat + (size_t)(b * Tc + crow) * Dd + h * HS + l16;
    #pragma unroll
    for (int e = 0; e < 8; ++e) {
        float* orow = obase + (size_t)e * Dd;
        orow[0]  = o0[e] * invl[e];
        orow[16] = o1[e] * invl[e];
        orow[32] = o2[e] * invl[e];
        orow[48] = o3[e] * invl[e];
    }
}

// ---------------------------------------------------------------------------
// Loss = -mean(log_softmax(logits)[targets]); one block per row of V=50257.
// ---------------------------------------------------------------------------
__global__ void zero_loss(float* loss) { if (threadIdx.x == 0) loss[0] = 0.f; }

__global__ __launch_bounds__(256)
void loss_kernel(const float* __restrict__ logits, const int* __restrict__ targets,
                 float* loss)
{
    int row = blockIdx.x, tid = threadIdx.x;
    const float* lr = logits + (size_t)row * Vv;
    __shared__ float red[256];

    float mx = -1e30f;
    for (int j = tid; j < Vv; j += 256) mx = fmaxf(mx, lr[j]);
    red[tid] = mx; __syncthreads();
    for (int off = 128; off; off >>= 1) {
        if (tid < off) red[tid] = fmaxf(red[tid], red[tid + off]);
        __syncthreads();
    }
    mx = red[0]; __syncthreads();

    float s = 0.f;
    for (int j = tid; j < Vv; j += 256) s += __expf(lr[j] - mx);
    red[tid] = s; __syncthreads();
    for (int off = 128; off; off >>= 1) {
        if (tid < off) red[tid] += red[tid + off];
        __syncthreads();
    }
    if (tid == 0) {
        float lse = mx + logf(red[0]);
        float lp  = lr[targets[row]] - lse;
        atomicAdd(loss, -lp / (float)BT);
    }
}

// ---------------------------------------------------------------------------
extern "C" void kernel_launch(void* const* d_in, const int* in_sizes, int n_in,
                              void* d_out, int out_size, void* d_ws, size_t ws_size,
                              hipStream_t stream)
{
    (void)in_sizes; (void)n_in; (void)out_size; (void)ws_size;
    const int*   idx     = (const int*)  d_in[0];
    const int*   targets = (const int*)  d_in[1];
    const float* tok_emb = (const float*)d_in[2];
    const float* pos_emb = (const float*)d_in[3];
    const float* wq      = (const float*)d_in[4];
    const float* wk      = (const float*)d_in[5];
    const float* wv      = (const float*)d_in[6];
    const float* wo      = (const float*)d_in[7];
    const float* bo      = (const float*)d_in[8];
    const float* ln1_s   = (const float*)d_in[9];
    const float* ln1_b   = (const float*)d_in[10];
    const float* ln2_s   = (const float*)d_in[11];
    const float* ln2_b   = (const float*)d_in[12];
    const float* w1      = (const float*)d_in[13];
    const float* b1      = (const float*)d_in[14];
    const float* w2      = (const float*)d_in[15];
    const float* b2      = (const float*)d_in[16];
    const float* lnf_s   = (const float*)d_in[17];
    const float* lnf_b   = (const float*)d_in[18];
    const float* head_w  = (const float*)d_in[19];
    const float* head_b  = (const float*)d_in[20];
    float* out = (float*)d_out;

    // workspace carve-up (~126 MB)
    char* p = (char*)d_ws;
    float* x    = (float*)p; p += (size_t)BT * Dd * sizeof(float);
    float* hbuf = (float*)p; p += (size_t)BT * Dd * sizeof(float);
    float* qb   = (float*)p; p += (size_t)Hh * BT * HS * sizeof(float);
    float* kbuf = (float*)p; p += (size_t)Hh * BT * HS * sizeof(float);
    float* vbuf = (float*)p; p += (size_t)Hh * BT * HS * sizeof(float);
    float* ocat = (float*)p; p += (size_t)BT * Dd * sizeof(float);
    float* mid  = (float*)p; p += (size_t)BT * D4 * sizeof(float);

    embed_kernel<<<BT, 256, 0, stream>>>(idx, tok_emb, pos_emb, x);

    const long long wqkvStride = (long long)Dd * HS;      // per-head weight
    const long long qkvCStride = (long long)BT * HS;      // per-head output

    for (int l = 0; l < Lc; ++l) {
        ln_kernel<<<BT, 256, 0, stream>>>(x, ln1_s + l * Dd, ln1_b + l * Dd, hbuf);

        dim3 gq(1, BT / 128, Hh);          // N=64 -> one 64-wide block column
        gemm_wmma<HS, Dd, false><<<gq, 256, 0, stream>>>(
            hbuf, wq + (size_t)l * Hh * Dd * HS, qb, nullptr, nullptr,
            wqkvStride, qkvCStride);
        gemm_wmma<HS, Dd, false><<<gq, 256, 0, stream>>>(
            hbuf, wk + (size_t)l * Hh * Dd * HS, kbuf, nullptr, nullptr,
            wqkvStride, qkvCStride);
        gemm_wmma<HS, Dd, false><<<gq, 256, 0, stream>>>(
            hbuf, wv + (size_t)l * Hh * Dd * HS, vbuf, nullptr, nullptr,
            wqkvStride, qkvCStride);

        // 16-query tiles: T/16 = 128 per (b,h); 4 waves per block
        attn_wmma_kernel<<<dim3(Tc / 64, Hh, Bc), dim3(32, 4), 0, stream>>>(
            qb, kbuf, vbuf, ocat);

        dim3 go(Dd / 64, BT / 128, 1);
        gemm_wmma<Dd, Dd, false><<<go, 256, 0, stream>>>(
            ocat, wo + (size_t)l * Dd * Dd, x, bo + l * Dd, x, 0, 0);

        ln_kernel<<<BT, 256, 0, stream>>>(x, ln2_s + l * Dd, ln2_b + l * Dd, hbuf);

        dim3 g1(D4 / 64, BT / 128, 1);
        gemm_wmma<D4, Dd, true><<<g1, 256, 0, stream>>>(
            hbuf, w1 + (size_t)l * Dd * D4, mid, b1 + l * D4, nullptr, 0, 0);

        dim3 g2(Dd / 64, BT / 128, 1);
        gemm_wmma<Dd, D4, false><<<g2, 256, 0, stream>>>(
            mid, w2 + (size_t)l * D4 * Dd, x, b2 + l * Dd, x, 0, 0);
    }

    ln_kernel<<<BT, 256, 0, stream>>>(x, lnf_s, lnf_b, hbuf);

    dim3 gh((Vv + 63) / 64, BT / 128, 1);                // 786 ragged-N blocks
    gemm_wmma<Vv, Dd, false><<<gh, 256, 0, stream>>>(
        hbuf, head_w, out, head_b, nullptr, 0, 0);

    float* loss = out + (size_t)BT * Vv;
    zero_loss<<<1, 32, 0, stream>>>(loss);
    loss_kernel<<<BT, 256, 0, stream>>>(out, targets, loss);
}
